// LSTMODEModel_58299886076006
// MI455X (gfx1250) — matrix operand order repaired and checked by reading, not compiled, not run on hardware
//
#include <hip/hip_runtime.h>

// ---------------------------------------------------------------------------
// Problem constants
// ---------------------------------------------------------------------------
#define BB   256
#define SS   512
#define DD   20
#define HH   128
#define DPAD 32
#define G4H  512          // 4*H
#define ROWS (BB * SS)    // 131072
#define SDTF 1.0e-4f      // DT / ODE_STEPS

typedef __attribute__((ext_vector_type(16))) _Float16    v16h;
typedef __attribute__((ext_vector_type(8)))  float       v8f;
typedef __attribute__((ext_vector_type(4)))  unsigned int v4u;

union F16x16 { v16h h; v4u u[2]; };

// ---------------------------------------------------------------------------
// WMMA fragment loaders (layouts per CDNA5 ISA 7.12.2, wave32).
// Callers precompute a per-lane base pointer once; per-fragment offsets are
// compile-time constants after unrolling -> loads become b128 with imm offsets.
//   A (16x32 f16): base = A + (lane&15)*lda + ((lane>>4)<<3); halves at +0, +16
//   B (32x16 f16): packed Wp[N][Kpad]; base = Wp + (n0+(lane&15))*ldk
//                  + ((lane>>4)<<3); halves at +0, +16
//   C/D (16x16 f32): m = r + 8*(lane>=16), n = lane&15
// ---------------------------------------------------------------------------
__device__ __forceinline__ v16h frag_ld_a(const _Float16* p) {
    F16x16 r;
    r.u[0] = *(const v4u*)(p);        // K kh..kh+7
    r.u[1] = *(const v4u*)(p + 16);   // K 16+kh..16+kh+7
    return r.h;
}

__device__ __forceinline__ v16h frag_ld_b(const _Float16* p) {
    F16x16 r;
    r.u[0] = *(const v4u*)(p);
    r.u[1] = *(const v4u*)(p + 16);
    return r.h;
}

#define WMMA_F16(a, b, c) \
    __builtin_amdgcn_wmma_f32_16x16x32_f16(false, (a), false, (b), (short)0, (c), false, false)

// ---------------------------------------------------------------------------
// Fast transcendentals: gfx1250 has HW tanh; build sigmoid from v_exp/v_rcp
// instead of branchy libm expansions (these sit on the serial scan's
// critical path).
// ---------------------------------------------------------------------------
__device__ __forceinline__ float fexp2_(float x) { return __builtin_amdgcn_exp2f(x); }
__device__ __forceinline__ float frcp_(float x)  { return __builtin_amdgcn_rcpf(x); }

__device__ __forceinline__ float fast_sigmoid(float x) {
    // 1 / (1 + exp(-x)) with exp(-x) = exp2(-x*log2 e)
    return frcp_(1.0f + fexp2_(-1.4426950408889634f * x));
}

#if __has_builtin(__builtin_amdgcn_tanhf)
__device__ __forceinline__ float fast_tanh(float x) { return __builtin_amdgcn_tanhf(x); }
#else
__device__ __forceinline__ float fast_tanh(float x) {
    // tanh(x) = (e^{2x}-1)/(e^{2x}+1), e^{2x} = exp2(2x*log2 e)
    float e = fexp2_(2.8853900817779268f * x);
    return (e - 1.0f) * frcp_(e + 1.0f);
}
#endif

// ---------------------------------------------------------------------------
// Generic f32 -> f16 pack: dst[r][c] = (c < ncols) ? src[r][col0 + c] : 0
// ---------------------------------------------------------------------------
__global__ void pack_f16_kernel(const float* __restrict__ src, _Float16* __restrict__ dst,
                                int rows, int src_ld, int col0, int ncols, int dst_ld) {
    long total  = (long)rows * dst_ld;
    long stride = (long)gridDim.x * blockDim.x;
    for (long i = (long)blockIdx.x * blockDim.x + threadIdx.x; i < total; i += stride) {
        int r = (int)(i / dst_ld);
        int c = (int)(i - (long)r * dst_ld);
        float v = (c < ncols) ? src[(long)r * src_ld + col0 + c] : 0.0f;
        dst[i] = (_Float16)v;
    }
}

// ---------------------------------------------------------------------------
// Fused LSTM recurrence. One workgroup = 16 batch rows, full 512-step scan.
// 8 waves; wave w owns gate columns [w*64, w*64+64).
// Input projection (K = KINPAD) is fused into each step.
// h lives in LDS (f16, feeds next step's A-frags); c lives in registers.
// Weight B-fragments are hoisted into VGPRs for the whole scan; all LDS /
// global addresses are precomputed per-lane so the serial loop is just
// wmma + ds ops + HW transcendentals.
// ---------------------------------------------------------------------------
template <int KINPAD>
__global__ __launch_bounds__(256)
void lstm_rec_kernel(const _Float16* __restrict__ xin,  // (B, S, KINPAD), row = b*S+s
                     const _Float16* __restrict__ wih,  // (512, KINPAD) packed
                     const _Float16* __restrict__ whh,  // (512, 128)    packed
                     const float*    __restrict__ b_ih,
                     const float*    __restrict__ b_hh,
                     _Float16* __restrict__ hout16,     // (B, S, 128) f16 or null
                     float*    __restrict__ hout32) {   // (B, S, 128) f32 or null
    constexpr int KI = KINPAD / 32;

    __shared__ _Float16 lds_h[16 * 128];       // 4 KB
    __shared__ float    lds_gates[16 * 512];   // 32 KB

    const int tid  = threadIdx.x;
    const int lane = tid & 31;
    const int wv   = tid >> 5;
    const int b0   = blockIdx.x * 16;

    for (int i = tid; i < 16 * 128; i += 256) lds_h[i] = (_Float16)0.0f;

    float c_reg[8];
#pragma unroll
    for (int e = 0; e < 8; ++e) c_reg[e] = 0.0f;

    // ---- per-lane fragment bases (loop-invariant) ----
    const int fm     = lane & 15;
    const int fa_off = (lane >> 4) << 3;   // A: K half-select
    const int fb_off = (lane >> 4) << 3;   // B: K half-select
    const int mB     = (lane >> 4) << 3;   // C/D row base

    // A from global input (advanced by KINPAD per step)
    const _Float16* aG = xin + ((long)(b0 + fm) * SS) * KINPAD + fa_off;
    // A from LDS h
    const _Float16* aH = (const _Float16*)lds_h + fm * 128 + fa_off;
    // B fragment bases
    const _Float16* bih_base = wih + (wv * 64 + fm) * KINPAD + fb_off;
    const _Float16* bhh_base = whh + (wv * 64 + fm) * 128 + fb_off;
    // gate store base (column = wv*64 + j*16 + fm, row = mB + r)
    float* gout = lds_gates + mB * 512 + wv * 64 + fm;
    // per-thread store-column biases (loop-invariant)
    float biasj[4];
#pragma unroll
    for (int j = 0; j < 4; ++j) {
        int n = wv * 64 + j * 16 + fm;
        biasj[j] = b_ih[n] + b_hh[n];
    }

    // ---- activation-phase bases: thread owns 8 consecutive h columns ----
    const int am  = tid >> 4;          // row m (0..15)
    const int ac  = (tid * 8) & 127;   // first h column
    const float*    gact = lds_gates + am * 512 + ac;
    _Float16*       hact = lds_h + am * 128 + ac;
    _Float16* o16 = hout16 ? hout16 + ((long)(b0 + am) * SS) * 128 + ac : (_Float16*)nullptr;
    float*    o32 = hout32 ? hout32 + ((long)(b0 + am) * SS) * 128 + ac : (float*)nullptr;

    // ---- hoist weight fragments into registers (reused 512x) ----
    v16h bin[4 * KI];
    v16h bhh[16];
#pragma unroll
    for (int j = 0; j < 4; ++j) {
#pragma unroll
        for (int kk = 0; kk < KI; ++kk)
            bin[j * KI + kk] = frag_ld_b(bih_base + j * 16 * KINPAD + kk * 32);
#pragma unroll
        for (int kk = 0; kk < 4; ++kk)
            bhh[j * 4 + kk] = frag_ld_b(bhh_base + j * 16 * 128 + kk * 32);
    }
    __syncthreads();

    for (int t = 0; t < SS; ++t) {
        // prefetch input rows two steps ahead (global_prefetch_b8)
        __builtin_prefetch(aG + 2 * KINPAD, 0, 1);

#pragma unroll
        for (int j = 0; j < 4; ++j) {
            v8f acc = {};
#pragma unroll
            for (int kk = 0; kk < KI; ++kk)               // fused input projection
                acc = WMMA_F16(frag_ld_a(aG + kk * 32), bin[j * KI + kk], acc);
#pragma unroll
            for (int kk = 0; kk < 4; ++kk)                // recurrent h @ w_hh^T
                acc = WMMA_F16(frag_ld_a(aH + kk * 32), bhh[j * 4 + kk], acc);
            float bb = biasj[j];
#pragma unroll
            for (int r = 0; r < 8; ++r)
                gout[r * 512 + j * 16] = acc[r] + bb;     // imm-offset ds_store
        }
        __syncthreads();

        // Gate nonlinearity + state update: 2048 elements / 256 threads.
#pragma unroll
        for (int e = 0; e < 8; ++e) {
            float gi = gact[e];
            float gf = gact[128 + e];
            float gg = gact[256 + e];
            float go = gact[384 + e];
            float c  = fast_sigmoid(gf) * c_reg[e] + fast_sigmoid(gi) * fast_tanh(gg);
            float h  = fast_sigmoid(go) * fast_tanh(c);
            c_reg[e] = c;
            hact[e]  = (_Float16)h;
            if (o16) o16[e] = (_Float16)h;
            if (o32) o32[e] = h;
        }
        __syncthreads();

        aG += KINPAD;
        if (o16) o16 += 128;
        if (o32) o32 += 128;
    }
}

// ---------------------------------------------------------------------------
// Fused ODE Euler + final projection. One workgroup = 16 rows of (B*S).
//   xc = x @ w1x^T + b1   (loop-invariant, computed once into LDS)
//   10x: mid = tanh(h @ w1h^T + xc); dh = tanh(mid @ w2^T + b2);
//        h = tanh(h + sdt*dh)
//   out = (0.5*lstm + 0.5*h) @ w_dir^T + b_dir
// w1h/w2 B-fragments register-resident for all 10 steps.
// ---------------------------------------------------------------------------
__global__ __launch_bounds__(256)
void ode_kernel(const _Float16* __restrict__ x16,    // (ROWS, 32) f16
                const float*    __restrict__ lstm32, // (ROWS, 128) f32
                const _Float16* __restrict__ w1h,    // (256, 128) packed
                const _Float16* __restrict__ w1x,    // (256, 32)  packed
                const float*    __restrict__ b1,     // (256)
                const _Float16* __restrict__ w2,     // (128, 256) packed
                const float*    __restrict__ b2,     // (128)
                const float*    __restrict__ wdir,   // (3, 128) f32
                const float*    __restrict__ bdir,   // (3)
                float* __restrict__ out) {           // (ROWS, 3)
    __shared__ _Float16 lds_h[16 * 128];     // 4 KB
    __shared__ _Float16 lds_mid[16 * 256];   // 8 KB
    __shared__ float    lds_xc[16 * 256];    // 16 KB

    const int  tid  = threadIdx.x;
    const int  lane = tid & 31;
    const int  wv   = tid >> 5;
    const long row0 = (long)blockIdx.x * 16;

    // h_ode <- lstm_out (f32 -> f16)
    for (int i = tid; i < 16 * 128; i += 256)
        lds_h[i] = (_Float16)lstm32[row0 * 128 + i];

    // ---- per-lane fragment bases ----
    const int fm     = lane & 15;
    const int fa_off = (lane >> 4) << 3;
    const int fb_off = (lane >> 4) << 3;
    const int mB     = (lane >> 4) << 3;

    const _Float16* aX = x16 + (row0 + fm) * 32 + fa_off;
    const _Float16* aH = (const _Float16*)lds_h   + fm * 128 + fa_off;
    const _Float16* aM = (const _Float16*)lds_mid + fm * 256 + fa_off;
    const _Float16* b1h_base = w1h + (wv * 32 + fm) * 128 + fb_off;
    const _Float16* b1x_base = w1x + (wv * 32 + fm) * 32 + fb_off;
    const _Float16* b2_base  = w2  + (wv * 16 + fm) * 256 + fb_off;

    // D/store bases
    float*    xcout  = lds_xc  + mB * 256 + wv * 32 + fm;  // [r*256 + j*16]
    _Float16* midout = lds_mid + mB * 256 + wv * 32 + fm;  // [r*256 + j*16]
    _Float16* hup    = lds_h   + mB * 128 + wv * 16 + fm;  // [r*128]
    const float b2v  = b2[wv * 16 + fm];
    float b1v[2];
#pragma unroll
    for (int j = 0; j < 2; ++j) b1v[j] = b1[wv * 32 + j * 16 + fm];

    // ---- hoist ODE weight fragments (reused 10x) ----
    v16h bf1[8], bf2[8];
#pragma unroll
    for (int j = 0; j < 2; ++j)
#pragma unroll
        for (int kk = 0; kk < 4; ++kk)
            bf1[j * 4 + kk] = frag_ld_b(b1h_base + j * 16 * 128 + kk * 32);
#pragma unroll
    for (int kk = 0; kk < 8; ++kk)
        bf2[kk] = frag_ld_b(b2_base + kk * 32);

    // xc = x @ w1x^T + b1  (invariant part of first ODE layer)
#pragma unroll
    for (int j = 0; j < 2; ++j) {
        v8f acc = {};
        acc = WMMA_F16(frag_ld_a(aX), frag_ld_b(b1x_base + j * 16 * 32), acc);
#pragma unroll
        for (int r = 0; r < 8; ++r)
            xcout[r * 256 + j * 16] = acc[r] + b1v[j];
    }
    __syncthreads();

    for (int s = 0; s < 10; ++s) {
        // mid = tanh(h @ w1h^T + xc); wave owns 32 of 256 cols
#pragma unroll
        for (int j = 0; j < 2; ++j) {
            v8f acc = {};
#pragma unroll
            for (int kk = 0; kk < 4; ++kk)
                acc = WMMA_F16(frag_ld_a(aH + kk * 32), bf1[j * 4 + kk], acc);
#pragma unroll
            for (int r = 0; r < 8; ++r)
                midout[r * 256 + j * 16] =
                    (_Float16)fast_tanh(acc[r] + xcout[r * 256 + j * 16]);
        }
        __syncthreads();

        // dh = tanh(mid @ w2^T + b2); h = tanh(h + sdt*dh); wave owns 16 of 128
        {
            v8f acc = {};
#pragma unroll
            for (int kk = 0; kk < 8; ++kk)
                acc = WMMA_F16(frag_ld_a(aM + kk * 32), bf2[kk], acc);
#pragma unroll
            for (int r = 0; r < 8; ++r) {
                float dh  = fast_tanh(acc[r] + b2v);
                float hol = (float)hup[r * 128];
                hup[r * 128] = (_Float16)fast_tanh(hol + SDTF * dh);
            }
        }
        __syncthreads();
    }

    // out = (0.5*lstm + 0.5*h_ode) @ w_dir^T + b_dir  (48 dots of length 128)
    if (tid < 48) {
        int m = tid / 3, o = tid % 3;
        const float* lr = lstm32 + (row0 + m) * 128;
        const _Float16* hr = (const _Float16*)lds_h + m * 128;
        float acc = bdir[o];
#pragma unroll 4
        for (int k = 0; k < 128; ++k)
            acc += 0.5f * (lr[k] + (float)hr[k]) * wdir[o * 128 + k];
        out[(row0 + m) * 3 + o] = acc;
    }
}

// ---------------------------------------------------------------------------
// Host-side launch
// ---------------------------------------------------------------------------
static inline size_t alignup(size_t x) { return (x + 255) & ~(size_t)255; }

extern "C" void kernel_launch(void* const* d_in, const int* in_sizes, int n_in,
                              void* d_out, int out_size, void* d_ws, size_t ws_size,
                              hipStream_t stream) {
    const float* x      = (const float*)d_in[0];
    const float* w_ih0  = (const float*)d_in[1];
    const float* w_hh0  = (const float*)d_in[2];
    const float* b_ih0  = (const float*)d_in[3];
    const float* b_hh0  = (const float*)d_in[4];
    const float* w_ih1  = (const float*)d_in[5];
    const float* w_hh1  = (const float*)d_in[6];
    const float* b_ih1  = (const float*)d_in[7];
    const float* b_hh1  = (const float*)d_in[8];
    const float* ode_w1 = (const float*)d_in[9];
    const float* ode_b1 = (const float*)d_in[10];
    const float* ode_w2 = (const float*)d_in[11];
    const float* ode_b2 = (const float*)d_in[12];
    const float* w_dir  = (const float*)d_in[13];
    const float* b_dir  = (const float*)d_in[14];
    float* out = (float*)d_out;

    // Workspace carve-up (~110 MB total)
    char* p = (char*)d_ws;
    _Float16* x16    = (_Float16*)p; p += alignup((size_t)ROWS * DPAD * 2);   // 8 MB
    _Float16* h1f16  = (_Float16*)p; p += alignup((size_t)ROWS * HH * 2);     // 32 MB
    float*    lstm32 = (float*)p;    p += alignup((size_t)ROWS * HH * 4);     // 64 MB
    _Float16* p_wih0 = (_Float16*)p; p += alignup((size_t)G4H * DPAD * 2);
    _Float16* p_whh0 = (_Float16*)p; p += alignup((size_t)G4H * HH * 2);
    _Float16* p_wih1 = (_Float16*)p; p += alignup((size_t)G4H * HH * 2);
    _Float16* p_whh1 = (_Float16*)p; p += alignup((size_t)G4H * HH * 2);
    _Float16* p_w1h  = (_Float16*)p; p += alignup((size_t)256 * HH * 2);
    _Float16* p_w1x  = (_Float16*)p; p += alignup((size_t)256 * DPAD * 2);
    _Float16* p_w2   = (_Float16*)p; p += alignup((size_t)HH * 256 * 2);
    (void)in_sizes; (void)n_in; (void)out_size; (void)ws_size;

    // --- pack inputs / weights to f16 (K padded to multiples of 32) ---
    pack_f16_kernel<<<4096, 256, 0, stream>>>(x,      x16,    ROWS, DD,      0,   DD,  DPAD);
    pack_f16_kernel<<<64,   256, 0, stream>>>(w_ih0,  p_wih0, G4H,  DD,      0,   DD,  DPAD);
    pack_f16_kernel<<<256,  256, 0, stream>>>(w_hh0,  p_whh0, G4H,  HH,      0,   HH,  HH);
    pack_f16_kernel<<<256,  256, 0, stream>>>(w_ih1,  p_wih1, G4H,  HH,      0,   HH,  HH);
    pack_f16_kernel<<<256,  256, 0, stream>>>(w_hh1,  p_whh1, G4H,  HH,      0,   HH,  HH);
    pack_f16_kernel<<<128,  256, 0, stream>>>(ode_w1, p_w1h,  256,  HH + DD, 0,   HH,  HH);   // h-part
    pack_f16_kernel<<<32,   256, 0, stream>>>(ode_w1, p_w1x,  256,  HH + DD, HH,  DD,  DPAD); // x-part
    pack_f16_kernel<<<128,  256, 0, stream>>>(ode_w2, p_w2,   HH,   256,     0,   256, 256);

    // --- LSTM layer 0: x(pad 32) -> h1 (f16) ---
    lstm_rec_kernel<DPAD><<<BB / 16, 256, 0, stream>>>(
        x16, p_wih0, p_whh0, b_ih0, b_hh0, h1f16, (float*)nullptr);

    // --- LSTM layer 1: h1 -> lstm_out (f32) ---
    lstm_rec_kernel<HH><<<BB / 16, 256, 0, stream>>>(
        h1f16, p_wih1, p_whh1, b_ih1, b_hh1, (_Float16*)nullptr, lstm32);

    // --- fused ODE Euler + blend + final projection ---
    ode_kernel<<<ROWS / 16, 256, 0, stream>>>(
        x16, lstm32, p_w1h, p_w1x, ode_b1, p_w2, ode_b2, w_dir, b_dir, out);
}
